// Model_33921651704507
// MI455X (gfx1250) — matrix operand order repaired and checked by reading, not compiled, not run on hardware
//
#include <hip/hip_runtime.h>
#include <hip/hip_bf16.h>
#include <math.h>

// ---------------------------------------------------------------------------
// Fused RNN cell + output projection for MI455X (gfx1250, wave32, WMMA).
//   new_h = tanh(x @ Wx^T + h @ Wh^T + B)      (two 4096^3 GEMMs, fused)
//   out   = new_h @ Wout^T + Bout              (4096x1024x4096)
// fp32 -> bf16 conversion happens while staging tiles into LDS; math runs on
// v_wmma_f32_16x16x32_bf16 with f32 accumulation.  Double-buffered LDS with a
// software pipeline: global prefetch -> WMMA on current buffer -> convert +
// ds_store into next buffer -> single barrier per K-step.
// ---------------------------------------------------------------------------

typedef __attribute__((ext_vector_type(16))) __bf16 v16bf;
typedef __attribute__((ext_vector_type(8)))  __bf16 v8bf;
typedef __attribute__((ext_vector_type(4)))  __bf16 v4bf;
typedef __attribute__((ext_vector_type(8)))  float  v8f;

#define BATCH  4096
#define INPUT  4096
#define HIDDEN 4096
#define OUTPUT 1024

#define BM 128   // block tile M
#define BN 128   // block tile N
#define BK 32    // K step == bf16 WMMA K
#define TILE_ELEMS (BM * BK)

// ---------------------------------------------------------------------------
// Fast tanh: native v_tanh_f32 when the toolchain exposes it (CDNA5 lists
// TANH as a hardware transcendental), else branch-free exp2+rcp identity.
// ---------------------------------------------------------------------------
__device__ __forceinline__ float fast_tanh(float v) {
#if __has_builtin(__builtin_amdgcn_tanhf)
  return __builtin_amdgcn_tanhf(v);
#else
  // tanh(v) = sign(v) * (1 - e^{-2|v|}) / (1 + e^{-2|v|});  e^x = 2^{x*log2 e}
  const float t = __builtin_amdgcn_exp2f(-2.885390081777927f * fabsf(v));
  const float r = __builtin_amdgcn_rcpf(1.0f + t);
  return copysignf((1.0f - t) * r, v);
#endif
}

// ---------------------------------------------------------------------------
// Global prefetch into registers: 128x32 fp32 tile -> 4 float4 per thread.
// ---------------------------------------------------------------------------
__device__ __forceinline__ void fetch_tile_f32(float4* regs,
                                               const float* __restrict__ src,
                                               int row0, int k0, int ld, int tid) {
#pragma unroll
  for (int i = 0; i < 4; ++i) {
    const int idx = tid + i * 256;   // 0..1023
    const int r   = idx >> 3;        // 0..127
    const int c4  = idx & 7;         // float4 chunk within the 32-wide K slab
    regs[i] = *reinterpret_cast<const float4*>(
        src + (size_t)(row0 + r) * (size_t)ld + (size_t)k0 + (size_t)(c4 * 4));
  }
}

// Convert prefetched fp32 regs -> bf16 and store into LDS [128][32].
__device__ __forceinline__ void stage_tile_bf16(__bf16* lds, const float4* regs,
                                                int tid) {
#pragma unroll
  for (int i = 0; i < 4; ++i) {
    const int idx = tid + i * 256;
    const int r   = idx >> 3;
    const int c4  = idx & 7;
    v4bf t;
    t[0] = (__bf16)regs[i].x; t[1] = (__bf16)regs[i].y;
    t[2] = (__bf16)regs[i].z; t[3] = (__bf16)regs[i].w;
    *reinterpret_cast<v4bf*>(lds + r * BK + c4 * 4) = t;
  }
}

// Global prefetch of a bf16 tile (new_h in kernel 2): 2 uint4 per thread.
__device__ __forceinline__ void fetch_tile_bf16(uint4* regs,
                                                const __bf16* __restrict__ src,
                                                int row0, int k0, int ld, int tid) {
#pragma unroll
  for (int i = 0; i < 2; ++i) {
    const int idx = tid + i * 256;   // 0..511
    const int r   = idx >> 2;        // 0..127
    const int c8  = idx & 3;         // 8-bf16 chunk
    regs[i] = *reinterpret_cast<const uint4*>(
        src + (size_t)(row0 + r) * (size_t)ld + (size_t)k0 + (size_t)(c8 * 8));
  }
}

__device__ __forceinline__ void stage_tile_raw(__bf16* lds, const uint4* regs,
                                               int tid) {
#pragma unroll
  for (int i = 0; i < 2; ++i) {
    const int idx = tid + i * 256;
    const int r   = idx >> 2;
    const int c8  = idx & 3;
    *reinterpret_cast<uint4*>(lds + r * BK + c8 * 8) = regs[i];
  }
}

// ---------------------------------------------------------------------------
// LDS -> VGPR fragment loads matching the CDNA5 WMMA layouts.
// A (16x32, MxK): lanes 0-15 row=lane hold K 0-7 / 16-23; lanes 16-31 hold
// K 8-15 / 24-31.  Two aligned 16-byte ds reads per lane.
// ---------------------------------------------------------------------------
__device__ __forceinline__ v16bf load_afrag(const __bf16* lds, int mbase, int lane) {
  const int r  = mbase + (lane & 15);
  const int kb = (lane >> 4) * 8;
  const __bf16* p = lds + r * BK + kb;
  const v8bf lo = *reinterpret_cast<const v8bf*>(p);
  const v8bf hi = *reinterpret_cast<const v8bf*>(p + 16);
  return __builtin_shufflevector(lo, hi, 0, 1, 2, 3, 4, 5, 6, 7,
                                 8, 9, 10, 11, 12, 13, 14, 15);
}

// B (32x16, KxN): lane l holds column N=l&15; lanes 0-15 hold K 0-7 / 16-23,
// lanes 16-31 hold K 8-15 / 24-31 (same order as A).
// LDS tile stored [N][K] (weight rows are B columns, K contiguous).
__device__ __forceinline__ v16bf load_bfrag(const __bf16* lds, int nbase, int lane) {
  const int c  = nbase + (lane & 15);
  const int kb = (lane >> 4) * 8;
  const __bf16* p = lds + c * BK + kb;
  const v8bf lo = *reinterpret_cast<const v8bf*>(p);
  const v8bf hi = *reinterpret_cast<const v8bf*>(p + 16);
  return __builtin_shufflevector(lo, hi, 0, 1, 2, 3, 4, 5, 6, 7,
                                 8, 9, 10, 11, 12, 13, 14, 15);
}

// 8 WMMAs for one wave's 64x32 sub-tile on one K=32 slab.
__device__ __forceinline__ void mma_step(const __bf16* la, const __bf16* lb,
                                         int wm, int wn, int lane,
                                         v8f acc[4][2]) {
  v16bf af[4], bfr[2];
#pragma unroll
  for (int mi = 0; mi < 4; ++mi) af[mi] = load_afrag(la, wm + mi * 16, lane);
#pragma unroll
  for (int ni = 0; ni < 2; ++ni) bfr[ni] = load_bfrag(lb, wn + ni * 16, lane);
#pragma unroll
  for (int mi = 0; mi < 4; ++mi)
#pragma unroll
    for (int ni = 0; ni < 2; ++ni)
      acc[mi][ni] = __builtin_amdgcn_wmma_f32_16x16x32_bf16(
          false, af[mi], false, bfr[ni], (short)0, acc[mi][ni], false, false);
}

// ---------------------------------------------------------------------------
// Kernel 1: new_h = tanh(x@Wx^T + h@Wh^T + B), bf16 result to workspace.
// Flattened 256-step pipelined K loop over both GEMM phases.
// ---------------------------------------------------------------------------
__global__ __launch_bounds__(256)
void rnn_cell_wmma(const float* __restrict__ x,
                   const float* __restrict__ hid,
                   const float* __restrict__ Wx,
                   const float* __restrict__ Wh,
                   const float* __restrict__ Bias,
                   __bf16* __restrict__ newh) {
  __shared__ alignas(16) __bf16 lds_a[2 * TILE_ELEMS];
  __shared__ alignas(16) __bf16 lds_b[2 * TILE_ELEMS];

  const int tid  = threadIdx.x;
  const int lane = tid & 31;
  const int wid  = tid >> 5;
  const int wm   = (wid & 1) * 64;
  const int wn   = (wid >> 1) * 32;
  const int m0   = blockIdx.x * BM;
  const int n0   = blockIdx.y * BN;

  v8f acc[4][2];
#pragma unroll
  for (int mi = 0; mi < 4; ++mi)
#pragma unroll
    for (int ni = 0; ni < 2; ++ni)
#pragma unroll
      for (int v = 0; v < 8; ++v) acc[mi][ni][v] = 0.0f;

  const int KSTEPS = INPUT / BK;        // 128 per phase
  const int NSTEP  = 2 * KSTEPS;        // 256 total

  float4 pa[4], pb[4];
  fetch_tile_f32(pa, x,  m0, 0, INPUT, tid);
  fetch_tile_f32(pb, Wx, n0, 0, INPUT, tid);
  stage_tile_bf16(lds_a, pa, tid);
  stage_tile_bf16(lds_b, pb, tid);
  __syncthreads();

#pragma unroll 1
  for (int s = 0; s < NSTEP; ++s) {
    const int cur  = s & 1;
    const int nxt  = cur ^ 1;
    const bool more = (s + 1) < NSTEP;

    if (more) {                                   // prefetch next slab
      const int s1 = s + 1;
      const float* A  = (s1 < KSTEPS) ? x  : hid;
      const float* Bw = (s1 < KSTEPS) ? Wx : Wh;
      const int k0 = (s1 & (KSTEPS - 1)) * BK;
      fetch_tile_f32(pa, A,  m0, k0, INPUT, tid);
      fetch_tile_f32(pb, Bw, n0, k0, INPUT, tid);
    }

    mma_step(lds_a + cur * TILE_ELEMS, lds_b + cur * TILE_ELEMS,
             wm, wn, lane, acc);

    if (more) {                                   // stage into other buffer
      stage_tile_bf16(lds_a + nxt * TILE_ELEMS, pa, tid);
      stage_tile_bf16(lds_b + nxt * TILE_ELEMS, pb, tid);
    }
    __syncthreads();
  }

  // Epilogue: bias + tanh, bf16 store.  C/D map: vgpr v, lane l ->
  // row = v + 8*(l>>4), col = l&15.
  const int rbase = m0 + wm;
  const int cbase = n0 + wn;
#pragma unroll
  for (int mi = 0; mi < 4; ++mi) {
#pragma unroll
    for (int ni = 0; ni < 2; ++ni) {
      const int c  = cbase + ni * 16 + (lane & 15);
      const float bb = Bias[c];
#pragma unroll
      for (int v = 0; v < 8; ++v) {
        const int r = rbase + mi * 16 + v + ((lane >> 4) << 3);
        newh[(size_t)r * HIDDEN + c] = (__bf16)fast_tanh(acc[mi][ni][v] + bb);
      }
    }
  }
}

// ---------------------------------------------------------------------------
// Kernel 2: out = new_h @ Wout^T + Bout (fp32 result).
// ---------------------------------------------------------------------------
__global__ __launch_bounds__(256)
void out_proj_wmma(const __bf16* __restrict__ newh,
                   const float* __restrict__ Wout,
                   const float* __restrict__ Bout,
                   float* __restrict__ out) {
  __shared__ alignas(16) __bf16 lds_a[2 * TILE_ELEMS];
  __shared__ alignas(16) __bf16 lds_b[2 * TILE_ELEMS];

  const int tid  = threadIdx.x;
  const int lane = tid & 31;
  const int wid  = tid >> 5;
  const int wm   = (wid & 1) * 64;
  const int wn   = (wid >> 1) * 32;
  const int m0   = blockIdx.x * BM;
  const int n0   = blockIdx.y * BN;

  v8f acc[4][2];
#pragma unroll
  for (int mi = 0; mi < 4; ++mi)
#pragma unroll
    for (int ni = 0; ni < 2; ++ni)
#pragma unroll
      for (int v = 0; v < 8; ++v) acc[mi][ni][v] = 0.0f;

  const int NSTEP = HIDDEN / BK;        // 128

  uint4  pa[2];
  float4 pb[4];
  fetch_tile_bf16(pa, newh, m0, 0, HIDDEN, tid);
  fetch_tile_f32 (pb, Wout, n0, 0, HIDDEN, tid);
  stage_tile_raw (lds_a, pa, tid);
  stage_tile_bf16(lds_b, pb, tid);
  __syncthreads();

#pragma unroll 1
  for (int s = 0; s < NSTEP; ++s) {
    const int cur  = s & 1;
    const int nxt  = cur ^ 1;
    const bool more = (s + 1) < NSTEP;

    if (more) {
      const int k0 = (s + 1) * BK;
      fetch_tile_bf16(pa, newh, m0, k0, HIDDEN, tid);
      fetch_tile_f32 (pb, Wout, n0, k0, HIDDEN, tid);
    }

    mma_step(lds_a + cur * TILE_ELEMS, lds_b + cur * TILE_ELEMS,
             wm, wn, lane, acc);

    if (more) {
      stage_tile_raw (lds_a + nxt * TILE_ELEMS, pa, tid);
      stage_tile_bf16(lds_b + nxt * TILE_ELEMS, pb, tid);
    }
    __syncthreads();
  }

  const int rbase = m0 + wm;
  const int cbase = n0 + wn;
#pragma unroll
  for (int mi = 0; mi < 4; ++mi) {
#pragma unroll
    for (int ni = 0; ni < 2; ++ni) {
      const int c  = cbase + ni * 16 + (lane & 15);
      const float bb = Bout[c];
#pragma unroll
      for (int v = 0; v < 8; ++v) {
        const int r = rbase + mi * 16 + v + ((lane >> 4) << 3);
        out[(size_t)r * OUTPUT + c] = acc[mi][ni][v] + bb;
      }
    }
  }
}

// ---------------------------------------------------------------------------
// Launch: kernel1 -> d_ws (bf16 new_h, 32 MB) -> kernel2 -> d_out (fp32).
// ---------------------------------------------------------------------------
extern "C" void kernel_launch(void* const* d_in, const int* in_sizes, int n_in,
                              void* d_out, int out_size, void* d_ws, size_t ws_size,
                              hipStream_t stream) {
  const float* x    = (const float*)d_in[0];
  const float* hid  = (const float*)d_in[1];
  const float* Wx   = (const float*)d_in[2];
  const float* Wh   = (const float*)d_in[3];
  const float* Bias = (const float*)d_in[4];
  const float* Wout = (const float*)d_in[5];
  const float* Bout = (const float*)d_in[6];
  float* out        = (float*)d_out;
  __bf16* newh      = (__bf16*)d_ws;   // BATCH*HIDDEN bf16 = 32 MB scratch

  dim3 block(256);
  dim3 grid1(BATCH / BM, HIDDEN / BN);   // 32 x 32
  dim3 grid2(BATCH / BM, OUTPUT / BN);   // 32 x 8

  rnn_cell_wmma<<<grid1, block, 0, stream>>>(x, hid, Wx, Wh, Bias, newh);
  out_proj_wmma<<<grid2, block, 0, stream>>>(newh, Wout, Bout, out);
}
